// dMaSIFConv_48584670052601
// MI455X (gfx1250) — compile-verified, matrix operand order checked by reading, not run on hardware
//
#include <hip/hip_runtime.h>

typedef __attribute__((ext_vector_type(16))) _Float16 v16h;
typedef __attribute__((ext_vector_type(8)))  float    v8f;

#define NPTS   1600
#define CIN    16
#define HIDD   64
#define NH     8
#define DKH    8
#define CUTS   8
#define NRBF   8
#define RPROJ  4

// ---------------------------------------------------------------------------
// Generic row-major linear layer: out[r][c] = act(sum_k in[r][k]*w[c][k] + b[c])
// act(x) = x>=0 ? x : slope*x   (slope=1.0 -> identity, slope=0.2 -> leaky)
// ---------------------------------------------------------------------------
__global__ void k_linear(const float* __restrict__ in, const float* __restrict__ w,
                         const float* __restrict__ b, float* __restrict__ out,
                         int n, int cin, int cout, float slope) {
  int idx = blockIdx.x * blockDim.x + threadIdx.x;
  if (idx >= n * cout) return;
  int r = idx / cout, c = idx - r * cout;
  float acc = b[c];
  for (int k = 0; k < cin; ++k) acc += in[r * cin + k] * w[c * cin + k];
  out[idx] = acc >= 0.f ? acc : slope * acc;
}

// ---------------------------------------------------------------------------
// GroupNorm stats over groups of 16 channels x all rows (groups=4, C=64)
// stats[g] = mu, stats[4+g] = var
// ---------------------------------------------------------------------------
__global__ void k_gn_stats(const float* __restrict__ x, float* __restrict__ stats) {
  __shared__ float ss[256], ss2[256];
  int g = blockIdx.x, t = threadIdx.x;
  float s = 0.f, s2 = 0.f;
  for (int e = t; e < NPTS * 16; e += 256) {
    int r = e >> 4, ch = g * 16 + (e & 15);
    float v = x[r * HIDD + ch];
    s += v; s2 += v * v;
  }
  ss[t] = s; ss2[t] = s2; __syncthreads();
  for (int o = 128; o > 0; o >>= 1) {
    if (t < o) { ss[t] += ss[t + o]; ss2[t] += ss2[t + o]; }
    __syncthreads();
  }
  if (t == 0) {
    float inv = 1.0f / (float)(NPTS * 16);
    float mu = ss[0] * inv;
    stats[g] = mu;
    stats[4 + g] = ss2[0] * inv - mu * mu;
  }
}

__global__ void k_gn_apply(const float* __restrict__ x, const float* __restrict__ stats,
                           const float* __restrict__ gamma, const float* __restrict__ beta,
                           float* __restrict__ out) {
  int idx = blockIdx.x * blockDim.x + threadIdx.x;
  if (idx >= NPTS * HIDD) return;
  int ch = idx & 63, g = ch >> 4;
  out[idx] = (x[idx] - stats[g]) * rsqrtf(stats[4 + g] + 1e-5f) * gamma[ch] + beta[ch];
}

// ---------------------------------------------------------------------------
// Softmax stats pass: for each (i, head) compute row max m and denominator z
// of S[i,j] = (Q[i,s] . K[j,s]) / sqrt(8) over j.
// ---------------------------------------------------------------------------
__global__ void k_softmax_stats(const float* __restrict__ Q, const float* __restrict__ K,
                                float* __restrict__ m, float* __restrict__ z) {
  int idx = blockIdx.x * blockDim.x + threadIdx.x;
  if (idx >= NPTS * NH) return;
  int i = idx >> 3, h = idx & 7;
  const float scale = 0.35355339059327373f;
  float q[DKH];
#pragma unroll
  for (int d = 0; d < DKH; ++d) q[d] = Q[i * HIDD + h * DKH + d];
  float mm = -1e30f, zz = 0.f;
  for (int j = 0; j < NPTS; ++j) {
    const float* kj = K + j * HIDD + h * DKH;
    float s = 0.f;
#pragma unroll
    for (int d = 0; d < DKH; ++d) s += q[d] * kj[d];
    s *= scale;
    if (s > mm) { zz = zz * __expf(mm - s) + 1.0f; mm = s; }
    else        { zz += __expf(s - mm); }
  }
  m[idx] = mm; z[idx] = zz;
}

// ---------------------------------------------------------------------------
// Pairwise aggregation kernel (the hot loop). One wave32 per 16-row i-tile.
// Per j: build per-pair features X (3 local coords, ndot, 4 RBF proj),
// F1 = relu(X @ conv_w1^T) on VALU (K=8, lanes 0-15 carry rows),
// Fh = relu(F1 @ conv_w2^T) via 4x v_wmma_f32_16x16x32_f16 (K padded 8->32),
// then acc += softmax(S)*W * Fh * V[j].
// ---------------------------------------------------------------------------
__launch_bounds__(128)
__global__ void k_pair_agg(const float* __restrict__ pts, const float* __restrict__ nuvg,
                           const float* __restrict__ Q, const float* __restrict__ K,
                           const float* __restrict__ V,
                           const float* __restrict__ mrow, const float* __restrict__ zrow,
                           const float* __restrict__ conv_w1, const float* __restrict__ conv_b1,
                           const float* __restrict__ conv_w2, const float* __restrict__ conv_b2,
                           const float* __restrict__ rbf_log_sigma,
                           const float* __restrict__ rbf_a, const float* __restrict__ rbf_b,
                           const float* __restrict__ rbf_centers,
                           float* __restrict__ agg) {
  __shared__ float ET[4][NH][16];          // per-wave softmax*window weights [head][iLocal]
  __shared__ float sw1[64], sb1[CUTS], sra[RPROJ * NRBF], srb[RPROJ], scen[NRBF];

  const int tid  = threadIdx.x;
  const int lane = tid & 31;
  const int wav  = tid >> 5;
  const int tile = blockIdx.x * 4 + wav;   // 100 tiles total
  const int i0   = tile * 16;
  const int il   = lane & 15;              // local row / WMMA column index
  const int hi   = lane >> 4;              // 0 or 1 (half-wave)
  const int i    = i0 + il;

  // stage small constants to LDS
  if (tid < 64) sw1[tid] = conv_w1[tid];
  if (tid < CUTS)  sb1[tid] = conv_b1[tid];
  if (tid < RPROJ * NRBF) sra[tid] = rbf_a[tid];
  if (tid < RPROJ) srb[tid] = rbf_b[tid];
  if (tid < NRBF)  scen[tid] = rbf_centers[tid];
  __syncthreads();

  const float invs  = 1.0f / (1.4142135623730951f * 6.0f);   // 1/(sqrt(2)*RADIUS)
  const float scale = 0.35355339059327373f;                  // 1/sqrt(D_K)

  // per-i data (replicated on lane and lane+16)
  const float pix = pts[i * 3 + 0] * invs;
  const float piy = pts[i * 3 + 1] * invs;
  const float piz = pts[i * 3 + 2] * invs;
  float nv[9];
#pragma unroll
  for (int a = 0; a < 9; ++a) nv[a] = nuvg[i * 9 + a];

  // this lane owns heads hb..hb+3 for the softmax weight computation
  const int hb = hi * 4;
  float q[32];
#pragma unroll
  for (int d = 0; d < 32; ++d) q[d] = Q[i * HIDD + hb * DKH + d];
  float mh[4], zih[4];
#pragma unroll
  for (int t = 0; t < 4; ++t) {
    mh[t]  = mrow[i * NH + hb + t];
    zih[t] = 1.0f / zrow[i * NH + hb + t];
  }

  float sig = fmaxf(__expf(rbf_log_sigma[0]), 1e-6f);
  const float inv2s2 = 1.0f / (2.0f * sig * sig);

  // B fragments for Fh GEMM: B[K=c][N=ch], K padded 8->32 with zeros.
  // layout: col N = lane&15, element e -> K = e + 16*hi
  v16h b2[4];
  float cb2[4];
#pragma unroll
  for (int t = 0; t < 4; ++t) {
    int ch = t * 16 + il;
    cb2[t] = conv_b2[ch];
#pragma unroll
    for (int e = 0; e < 16; ++e) {
      int c  = e + 16 * hi;
      int cc = c < 7 ? c : 7;
      float wv = conv_w2[ch * CUTS + cc];
      b2[t][e] = (c < CUTS) ? (_Float16)wv : (_Float16)0.0f;
    }
  }

  v8f acc[4] = {};   // C/D layout: col = t*16+il, row = r + 8*hi

  for (int j = 0; j < NPTS; ++j) {
    // --- per-pair geometry (all lanes; lanes l and l+16 share the same i) ---
    const float pjx = pts[j * 3 + 0] * invs;
    const float pjy = pts[j * 3 + 1] * invs;
    const float pjz = pts[j * 3 + 2] * invs;
    const float njx = nuvg[j * 9 + 0], njy = nuvg[j * 9 + 1], njz = nuvg[j * 9 + 2];
    const float dx = pjx - pix, dy = pjy - piy, dz = pjz - piz;
    const float ndot = nv[0] * njx + nv[1] * njy + nv[2] * njz;
    const float r2 = dx * dx + dy * dy + dz * dz;
    const float fw = 2.0f - ndot;
    const float d2 = r2 * fw * fw;
    const float tw = 1.0f + d2 * (1.0f / 3.0f);
    const float W  = 1.0f / (tw * tw * tw);                  // (1+d2/a)^-a, a=3
    const float based = (d2 > 0.0f) ? sqrtf(d2) : 0.0f;

    // --- softmax*window weights for this lane's 4 heads -> LDS ---
    const float* kj = K + j * HIDD + hb * DKH;
#pragma unroll
    for (int t = 0; t < 4; ++t) {
      float s = 0.f;
#pragma unroll
      for (int d = 0; d < DKH; ++d) s += q[t * DKH + d] * kj[t * DKH + d];
      s *= scale;
      ET[wav][hb + t][il] = __expf(s - mh[t]) * zih[t] * W;
    }

    // --- A fragment: F1 (16 pairs x 8 cuts), rows live on lanes 0..15 ---
    v16h a = {};
    if (hi == 0) {
      float X[8];
      X[0] = nv[0] * dx + nv[1] * dy + nv[2] * dz;
      X[1] = nv[3] * dx + nv[4] * dy + nv[5] * dz;
      X[2] = nv[6] * dx + nv[7] * dy + nv[8] * dz;
      X[3] = ndot;
      float R[NRBF];
#pragma unroll
      for (int k = 0; k < NRBF; ++k) {
        float dd = based - scen[k];
        R[k] = __expf(-dd * dd * inv2s2);
      }
#pragma unroll
      for (int p = 0; p < RPROJ; ++p) {
        float rp = srb[p];
#pragma unroll
        for (int k = 0; k < NRBF; ++k) rp += R[k] * sra[p * NRBF + k];
        X[4 + p] = rp;
      }
#pragma unroll
      for (int c = 0; c < CUTS; ++c) {
        float f1 = sb1[c];
#pragma unroll
        for (int f = 0; f < 8; ++f) f1 += X[f] * sw1[c * 8 + f];
        a[c] = (_Float16)fmaxf(f1, 0.0f);
      }
    }
    __syncthreads();   // ET writes visible

    // --- Fh = relu(F1 @ conv_w2^T) via WMMA; fold into accumulator ---
#pragma unroll
    for (int t = 0; t < 4; ++t) {
      v8f c = {};
      c = __builtin_amdgcn_wmma_f32_16x16x32_f16(false, a, false, b2[t],
                                                 (short)0, c, false, false);
      const int ch   = t * 16 + il;
      const int head = t * 2 + (il >> 3);
      const float vj = V[j * HIDD + ch];
      const float* et = &ET[wav][head][hi * 8];
#pragma unroll
      for (int r = 0; r < 8; ++r) {
        float fh = fmaxf(c[r] + cb2[t], 0.0f);
        acc[t][r] += et[r] * fh * vj;
      }
    }
    __syncthreads();   // done reading ET before next j overwrites it
  }

  // writeback: agg[i0 + r + 8*hi][t*16 + il]
#pragma unroll
  for (int t = 0; t < 4; ++t)
#pragma unroll
    for (int r = 0; r < 8; ++r)
      agg[(i0 + r + 8 * hi) * HIDD + t * 16 + il] = acc[t][r];
}

// ---------------------------------------------------------------------------
// Final: group-norm(o2) + features @ w_res^T + b_res
// ---------------------------------------------------------------------------
__global__ void k_final(const float* __restrict__ o2, const float* __restrict__ stats,
                        const float* __restrict__ gamma, const float* __restrict__ beta,
                        const float* __restrict__ feat, const float* __restrict__ wres,
                        const float* __restrict__ bres, float* __restrict__ out) {
  int idx = blockIdx.x * blockDim.x + threadIdx.x;
  if (idx >= NPTS * HIDD) return;
  int r = idx >> 6, ch = idx & 63, g = ch >> 4;
  float v = (o2[idx] - stats[g]) * rsqrtf(stats[4 + g] + 1e-5f) * gamma[ch] + beta[ch];
  float res = bres[ch];
#pragma unroll
  for (int k = 0; k < CIN; ++k) res += feat[r * CIN + k] * wres[ch * CIN + k];
  out[idx] = v + res;
}

// ---------------------------------------------------------------------------
extern "C" void kernel_launch(void* const* d_in, const int* in_sizes, int n_in,
                              void* d_out, int out_size, void* d_ws, size_t ws_size,
                              hipStream_t stream) {
  const float* points  = (const float*)d_in[0];
  const float* nuv     = (const float*)d_in[1];
  const float* feat    = (const float*)d_in[2];
  const float* w_in1   = (const float*)d_in[3];
  const float* b_in1   = (const float*)d_in[4];
  const float* w_in2   = (const float*)d_in[5];
  const float* b_in2   = (const float*)d_in[6];
  const float* g_in    = (const float*)d_in[7];
  const float* be_in   = (const float*)d_in[8];
  const float* wq      = (const float*)d_in[9];
  const float* bq      = (const float*)d_in[10];
  const float* wk      = (const float*)d_in[11];
  const float* bk      = (const float*)d_in[12];
  const float* wv      = (const float*)d_in[13];
  const float* bv      = (const float*)d_in[14];
  const float* conv_w1 = (const float*)d_in[15];
  const float* conv_b1 = (const float*)d_in[16];
  const float* conv_w2 = (const float*)d_in[17];
  const float* conv_b2 = (const float*)d_in[18];
  const float* rls     = (const float*)d_in[19];
  const float* rbf_a   = (const float*)d_in[20];
  const float* rbf_b   = (const float*)d_in[21];
  const float* rbf_c   = (const float*)d_in[22];
  const float* w_o1    = (const float*)d_in[23];
  const float* b_o1    = (const float*)d_in[24];
  const float* w_o2    = (const float*)d_in[25];
  const float* b_o2    = (const float*)d_in[26];
  const float* g_out   = (const float*)d_in[27];
  const float* be_out  = (const float*)d_in[28];
  const float* w_res   = (const float*)d_in[29];
  const float* b_res   = (const float*)d_in[30];
  (void)in_sizes; (void)n_in; (void)out_size; (void)ws_size;

  const size_t NM = (size_t)NPTS * HIDD;
  float* ws   = (float*)d_ws;
  float* h1   = ws;            // 1600x64
  float* h2   = h1 + NM;
  float* h    = h2 + NM;
  float* Q    = h  + NM;
  float* Kb   = Q  + NM;
  float* Vb   = Kb + NM;
  float* mr   = Vb + NM;       // 1600x8
  float* zr   = mr + (size_t)NPTS * NH;
  float* agg  = zr + (size_t)NPTS * NH;
  float* o1   = agg + NM;
  float* o2   = o1 + NM;
  float* st1  = o2 + NM;       // 8
  float* st2  = st1 + 8;       // 8

  const int TPB = 256;
  const int nElem = NPTS * HIDD;
  const int gElem = (nElem + TPB - 1) / TPB;

  // input MLP + group norm
  k_linear<<<(NPTS * HIDD + TPB - 1) / TPB, TPB, 0, stream>>>(feat, w_in1, b_in1, h1, NPTS, CIN, HIDD, 0.2f);
  k_linear<<<gElem, TPB, 0, stream>>>(h1, w_in2, b_in2, h2, NPTS, HIDD, HIDD, 0.2f);
  k_gn_stats<<<4, 256, 0, stream>>>(h2, st1);
  k_gn_apply<<<gElem, TPB, 0, stream>>>(h2, st1, g_in, be_in, h);

  // Q, K, V projections (identity activation: slope = 1)
  k_linear<<<gElem, TPB, 0, stream>>>(h, wq, bq, Q,  NPTS, HIDD, HIDD, 1.0f);
  k_linear<<<gElem, TPB, 0, stream>>>(h, wk, bk, Kb, NPTS, HIDD, HIDD, 1.0f);
  k_linear<<<gElem, TPB, 0, stream>>>(h, wv, bv, Vb, NPTS, HIDD, HIDD, 1.0f);

  // softmax row stats
  k_softmax_stats<<<(NPTS * NH + TPB - 1) / TPB, TPB, 0, stream>>>(Q, Kb, mr, zr);

  // heavy pairwise pass (100 wave-tiles, 4 waves/block)
  k_pair_agg<<<25, 128, 0, stream>>>(points, nuv, Q, Kb, Vb, mr, zr,
                                     conv_w1, conv_b1, conv_w2, conv_b2,
                                     rls, rbf_a, rbf_b, rbf_c, agg);

  // output MLP + group norm + residual
  k_linear<<<gElem, TPB, 0, stream>>>(agg, w_o1, b_o1, o1, NPTS, HIDD, HIDD, 0.2f);
  k_linear<<<gElem, TPB, 0, stream>>>(o1,  w_o2, b_o2, o2, NPTS, HIDD, HIDD, 0.2f);
  k_gn_stats<<<4, 256, 0, stream>>>(o2, st2);
  k_final<<<gElem, TPB, 0, stream>>>(o2, st2, g_out, be_out, feat, w_res, b_res, (float*)d_out);
}
